// CausalSelfAttention_27882927685803
// MI455X (gfx1250) — matrix operand order repaired and checked by reading, not compiled, not run on hardware
//
#include <hip/hip_runtime.h>
#include <hip/hip_bf16.h>

// ---------------------------------------------------------------------------
// CausalSelfAttention for MI455X (gfx1250, wave32, WMMA bf16 + TDM)
//   B=2, T=2048, C=2048, H=16, D=128
// Pipeline:
//   1) x -> bf16 ; W_attn,W_proj -> bf16 transposed [N,K]
//   2) WMMA GEMM qkv = x @ W_attn (TDM double-buffered LDS staging),
//      scatter to q[BH,T,D], k[BH,T,D], vT[BH,D,T]
//   3) flash attention per 16-row q tile (1 wave), WMMA QK^T and PV
//   4) WMMA GEMM out = y @ W_proj (fp32 output)
// ---------------------------------------------------------------------------

typedef __attribute__((ext_vector_type(16))) __bf16 v16bf;
typedef __attribute__((ext_vector_type(8)))  float  v8f;
typedef __attribute__((ext_vector_type(4)))  unsigned int u32x4;
typedef __attribute__((ext_vector_type(8)))  int i32x8;
typedef __attribute__((ext_vector_type(4)))  int i32x4;

static constexpr int BATCH = 2;
static constexpr int T     = 2048;
static constexpr int C     = 2048;
static constexpr int H     = 16;
static constexpr int DH    = 128;   // head dim
static constexpr int BH    = BATCH * H;          // 32
static constexpr int MROWS = BATCH * T;          // 4096
static constexpr int N3    = 3 * C;              // 6144

union FragU { u32x4 u[2]; v16bf v; };

__device__ __forceinline__ v16bf load_frag(const void* p0, const void* p1) {
  FragU f;
  f.u[0] = *(const u32x4*)p0;
  f.u[1] = *(const u32x4*)p1;
  return f.v;
}

__device__ __forceinline__ v8f vzero8() {
  v8f z = {0.f, 0.f, 0.f, 0.f, 0.f, 0.f, 0.f, 0.f};
  return z;
}

__device__ __forceinline__ v8f wmma_bf16(v16bf a, v16bf b, v8f c) {
  // D(16x16 f32) = A(16x32 bf16) * B(32x16 bf16) + C
  return __builtin_amdgcn_wmma_f32_16x16x32_bf16(
      false, a, false, b, (short)0, c, false, false);
}

// ---------------------------------------------------------------------------
// Tensor Data Mover: 2D tile load  Global -> LDS  (CDNA5 D# descriptor).
//   tile: tile1 rows x tile0 elems (2-byte elements), row stride = strideElems.
//   Destination LDS: contiguous tile1*tile0 elements at lds_addr.
// Issued once per calling wave; tracked with TENSORcnt.
// ---------------------------------------------------------------------------
__device__ __forceinline__ void tdm_load_2d_bf16(unsigned lds_addr,
                                                 const void* gaddr,
                                                 unsigned tile0, unsigned tile1,
                                                 unsigned strideElems) {
  const unsigned long long ga = (unsigned long long)(uintptr_t)gaddr;
  u32x4 g0;
  g0.x = 1u;                                     // count=1, user descriptor
  g0.y = lds_addr;                               // LDS byte address
  g0.z = (unsigned)(ga & 0xFFFFFFFFu);           // global_addr[31:0]
  g0.w = (unsigned)((ga >> 32) & 0x01FFFFFFu)    // global_addr[56:32]
         | (2u << 30);                           // type=2 ("image")
  i32x8 g1;
  g1[0] = (int)(1u << 16);                       // data_size=1 (2 bytes/elem)
  g1[1] = (int)((tile0 & 0xFFFFu) << 16);        // tensor_dim0[15:0]
  g1[2] = (int)((tile0 >> 16) | ((tile1 & 0xFFFFu) << 16));  // dim0 hi | dim1 lo
  g1[3] = (int)((tile1 >> 16) | ((tile0 & 0xFFFFu) << 16));  // dim1 hi | tile_dim0
  g1[4] = (int)(tile1 & 0xFFFFu);                // tile_dim1 | tile_dim2=0
  g1[5] = (int)strideElems;                      // tensor_dim0_stride[31:0]
  g1[6] = 0;                                     // stride hi | dim1_stride lo
  g1[7] = 0;
  const i32x4 z4 = {0, 0, 0, 0};                 // groups 2/3 unused (2D)
  const i32x8 z8 = {0, 0, 0, 0, 0, 0, 0, 0};
  __builtin_amdgcn_tensor_load_to_lds(g0, g1, z4, z4, z8, 0);
}

// ---------------------------------------------------------------------------
// fp32 -> bf16 elementwise
// ---------------------------------------------------------------------------
__global__ __launch_bounds__(256) void cvt_f32_bf16_kernel(
    const float* __restrict__ in, __bf16* __restrict__ out, int n) {
  int i = blockIdx.x * 256 + threadIdx.x;
  if (i < n) out[i] = (__bf16)in[i];
}

// ---------------------------------------------------------------------------
// fp32 [K,N] -> bf16 [N,K] (transpose through LDS tile)
// ---------------------------------------------------------------------------
__global__ __launch_bounds__(256) void transpose_f32_bf16_kernel(
    const float* __restrict__ in, __bf16* __restrict__ out, int K, int N) {
  __shared__ float tile[16][17];
  const int n0 = blockIdx.x * 16;
  const int k0 = blockIdx.y * 16;
  const int tx = threadIdx.x & 15;
  const int ty = threadIdx.x >> 4;
  tile[ty][tx] = in[(size_t)(k0 + ty) * N + (n0 + tx)];
  __syncthreads();
  out[(size_t)(n0 + ty) * K + (k0 + tx)] = (__bf16)tile[tx][ty];
}

// ---------------------------------------------------------------------------
// WMMA GEMM: C[M,N] = A[M,K](bf16,row-major) * Bt[N,K](bf16, N-major)
// Block: 256 threads (8 waves). Block tile 128x128, wave tile 32x64.
// Staging: TDM tensor_load_to_lds, double-buffered LDS (wave 0 issues,
// TENSORcnt-tracked, published to the workgroup by split barriers).
// mode 0: write fp32 C row-major
// mode 1: QKV scatter -> q[BH,T,DH], k[BH,T,DH], vT[BH,DH,T] (bf16)
// ---------------------------------------------------------------------------
__global__ __launch_bounds__(256) void gemm_bf16_wmma_kernel(
    const __bf16* __restrict__ A, const __bf16* __restrict__ Bt,
    float* __restrict__ Cf,
    __bf16* __restrict__ qh, __bf16* __restrict__ kh, __bf16* __restrict__ vth,
    int M, int N, int K, int mode) {
  __shared__ __align__(16) __bf16 Asub[2][128 * 32];
  __shared__ __align__(16) __bf16 Bsub[2][128 * 32];

  const int tid  = threadIdx.x;
  const int lane = tid & 31;
  const int hw   = (lane >> 4) & 1;   // half-wave
  const int ln   = lane & 15;
  const int wave = tid >> 5;
  const int wm   = (wave & 3) * 32;   // wave row offset in block tile
  const int wn   = (wave >> 2) * 64;  // wave col offset in block tile
  const int m0   = blockIdx.y * 128;
  const int n0   = blockIdx.x * 128;
  const bool issuer = (wave == 0);    // one wave drives the TDM

  v8f acc[2][4];
#pragma unroll
  for (int mi = 0; mi < 2; ++mi)
#pragma unroll
    for (int ni = 0; ni < 4; ++ni) acc[mi][ni] = vzero8();

  // ---- TDM stage of one K-step (A 128x32 tile + B 128x32 tile -> LDS)
  auto stage = [&](int buf, int kk) {
    if (!issuer) return;
    tdm_load_2d_bf16((unsigned)(uintptr_t)&Asub[buf][0],
                     A + (size_t)m0 * K + kk, 32u, 128u, (unsigned)K);
    tdm_load_2d_bf16((unsigned)(uintptr_t)&Bsub[buf][0],
                     Bt + (size_t)n0 * K + kk, 32u, 128u, (unsigned)K);
  };

  stage(0, 0);  // prologue: fill buffer 0

  for (int kk = 0; kk < K; kk += 32) {
    const int cur = (kk >> 5) & 1;
    if (kk + 32 < K) {
      stage(cur ^ 1, kk + 32);                       // prefetch next tile
      if (issuer) __builtin_amdgcn_s_wait_tensorcnt(2);  // current tile done
    } else {
      if (issuer) __builtin_amdgcn_s_wait_tensorcnt(0);
    }
    __syncthreads();  // publish TDM-written LDS to all waves

    // ---- build fragments (ISA A 16x32 / B 32x16 bf16 VGPR layouts)
    v16bf afr[2], bfr[4];
#pragma unroll
    for (int mi = 0; mi < 2; ++mi) {
      const char* p = (const char*)&Asub[cur][0] +
                      (size_t)(wm + mi * 16 + ln) * 64 + hw * 16;
      afr[mi] = load_frag(p, p + 32);
    }
#pragma unroll
    for (int ni = 0; ni < 4; ++ni) {
      const char* p = (const char*)&Bsub[cur][0] +
                      (size_t)(wn + ni * 16 + ln) * 64 + hw * 16;
      bfr[ni] = load_frag(p, p + 32);
    }
#pragma unroll
    for (int mi = 0; mi < 2; ++mi)
#pragma unroll
      for (int ni = 0; ni < 4; ++ni)
        acc[mi][ni] = wmma_bf16(afr[mi], bfr[ni], acc[mi][ni]);
    __syncthreads();  // all waves done reading buf[cur] before it is refilled
  }

  // ---- epilogue: C layout lane=col(ln), VGPR r = row r + hw*8
#pragma unroll
  for (int mi = 0; mi < 2; ++mi) {
#pragma unroll
    for (int ni = 0; ni < 4; ++ni) {
#pragma unroll
      for (int r = 0; r < 8; ++r) {
        const int row = m0 + wm + mi * 16 + r + hw * 8;
        const int col = n0 + wn + ni * 16 + ln;
        const float val = acc[mi][ni][r];
        if (mode == 0) {
          Cf[(size_t)row * N + col] = val;
        } else {
          const int b = row >> 11;        // / T
          const int t = row & (T - 1);
          const int part = col >> 11;     // / C : 0=q 1=k 2=v
          const int cc = col & (C - 1);
          const int h = cc >> 7;          // / DH
          const int d = cc & (DH - 1);
          const int bh = b * H + h;
          if (part == 0) {
            qh[((size_t)bh * T + t) * DH + d] = (__bf16)val;
          } else if (part == 1) {
            kh[((size_t)bh * T + t) * DH + d] = (__bf16)val;
          } else {
            vth[((size_t)bh * DH + d) * T + t] = (__bf16)val;
          }
        }
      }
    }
  }
}

// ---------------------------------------------------------------------------
// Flash attention: 1 wave per (bh, 16-row q tile). Keys in blocks of 32.
//   S(16x32) = Q(16x128) @ K^T   (8 wmma)
//   online softmax (row stats via __shfl_xor over 16-lane halves)
//   O(16x128) += P(16x32) @ V    (8 wmma), V stored transposed [DH,T]
// Output y written bf16 in [B*T, C] layout for the proj GEMM.
// ---------------------------------------------------------------------------
__global__ __launch_bounds__(32) void flash_attn_wmma_kernel(
    const __bf16* __restrict__ qh, const __bf16* __restrict__ kh,
    const __bf16* __restrict__ vth, __bf16* __restrict__ yb) {
  __shared__ __align__(16) __bf16 pbuf[16 * 32];

  const int lane = threadIdx.x & 31;
  const int hw   = lane >> 4;
  const int ln   = lane & 15;
  const int blk  = blockIdx.x;
  const int bh   = blk >> 7;              // / (T/16)
  const int qt   = (blk & 127) << 4;      // q tile start row
  const int b    = bh >> 4;
  const int h    = bh & 15;

  const char* qbase = (const char*)(qh  + (size_t)bh * T * DH);
  const char* kbase = (const char*)(kh  + (size_t)bh * T * DH);
  const char* vbase = (const char*)(vth + (size_t)bh * DH * T);

  // Q A-fragments: 4 chunks of K-dim 32 over DH=128
  v16bf qf[4];
  {
    const char* qrow = qbase + (size_t)(qt + ln) * (DH * 2);
#pragma unroll
    for (int c = 0; c < 4; ++c) {
      const char* p = qrow + c * 64 + hw * 16;
      qf[c] = load_frag(p, p + 32);
    }
  }

  v8f o[8];
#pragma unroll
  for (int n = 0; n < 8; ++n) o[n] = vzero8();
  float mrun[8], lrun[8];
#pragma unroll
  for (int r = 0; r < 8; ++r) { mrun[r] = -3.0e38f; lrun[r] = 0.f; }

  const float scale = 0.08838834764831845f;  // 1/sqrt(128)

  for (int kb = 0; kb <= qt; kb += 32) {
    // ---- S = Q @ K^T for 32 keys (two 16-col blocks)
    v8f s[2];
    s[0] = vzero8(); s[1] = vzero8();
#pragma unroll
    for (int j = 0; j < 2; ++j) {
      const char* krow = kbase + (size_t)(kb + j * 16 + ln) * (DH * 2);
#pragma unroll
      for (int c = 0; c < 4; ++c) {
        const char* p = krow + c * 64 + hw * 16;
        v16bf kf = load_frag(p, p + 32);
        s[j] = wmma_bf16(qf[c], kf, s[j]);
      }
    }
    // ---- scale + causal mask
#pragma unroll
    for (int j = 0; j < 2; ++j) {
#pragma unroll
      for (int r = 0; r < 8; ++r) {
        const int row = qt + r + hw * 8;
        const int col = kb + j * 16 + ln;
        const float v = s[j][r] * scale;
        s[j][r] = (col <= row) ? v : -3.0e38f;
      }
    }
    // ---- online softmax (rows live across 16 lanes of each half-wave)
#pragma unroll
    for (int r = 0; r < 8; ++r) {
      float mx = fmaxf(s[0][r], s[1][r]);
#pragma unroll
      for (int off = 8; off >= 1; off >>= 1)
        mx = fmaxf(mx, __shfl_xor(mx, off, 32));
      const float mnew  = fmaxf(mrun[r], mx);
      const float alpha = __expf(mrun[r] - mnew);
      const float p0 = __expf(s[0][r] - mnew);
      const float p1 = __expf(s[1][r] - mnew);
      float rs = p0 + p1;
#pragma unroll
      for (int off = 8; off >= 1; off >>= 1)
        rs += __shfl_xor(rs, off, 32);
      lrun[r] = lrun[r] * alpha + rs;
      mrun[r] = mnew;
#pragma unroll
      for (int n = 0; n < 8; ++n) o[n][r] *= alpha;
      // C-layout -> LDS (row r + hw*8, cols j*16+ln)
      pbuf[(r + hw * 8) * 32 + ln]      = (__bf16)p0;
      pbuf[(r + hw * 8) * 32 + 16 + ln] = (__bf16)p1;
    }
    // ---- reload P as A-fragment (16x32 bf16)
    const char* pl = (const char*)pbuf + (size_t)ln * 64 + hw * 16;
    v16bf pf = load_frag(pl, pl + 32);
    // ---- O += P @ V  (V^T rows are d, contiguous in keys)
#pragma unroll
    for (int n = 0; n < 8; ++n) {
      const char* vrow = vbase + (size_t)(n * 16 + ln) * (T * 2) + kb * 2 + hw * 16;
      v16bf vf = load_frag(vrow, vrow + 32);
      o[n] = wmma_bf16(pf, vf, o[n]);
    }
  }

  // ---- normalize and write y (bf16, [B*T, C])
#pragma unroll
  for (int n = 0; n < 8; ++n) {
#pragma unroll
    for (int r = 0; r < 8; ++r) {
      const int t = qt + r + hw * 8;
      const int d = n * 16 + ln;
      const float val = o[n][r] / lrun[r];
      yb[((size_t)(b * T + t)) * C + h * DH + d] = (__bf16)val;
    }
  }
}

// ---------------------------------------------------------------------------
extern "C" void kernel_launch(void* const* d_in, const int* in_sizes, int n_in,
                              void* d_out, int out_size, void* d_ws, size_t ws_size,
                              hipStream_t stream) {
  const float* x  = (const float*)d_in[0];   // [B,T,C]
  const float* Wa = (const float*)d_in[1];   // [C,3C]
  const float* Wp = (const float*)d_in[2];   // [C,C]
  float* out = (float*)d_out;                // [B,T,C]

  char* ws = (char*)d_ws;
  auto carve = [&](size_t bytes) {
    char* p = ws;
    ws += (bytes + 255) & ~(size_t)255;
    return p;
  };
  __bf16* xb  = (__bf16*)carve((size_t)MROWS * C  * 2);  // x bf16
  __bf16* WaT = (__bf16*)carve((size_t)N3    * C  * 2);  // W_attn^T bf16 [N3,C]
  __bf16* WpT = (__bf16*)carve((size_t)C     * C  * 2);  // W_proj^T bf16 [C,C]
  __bf16* qhB = (__bf16*)carve((size_t)BH * T * DH * 2); // q [BH,T,DH]
  __bf16* khB = (__bf16*)carve((size_t)BH * T * DH * 2); // k [BH,T,DH]
  __bf16* vtB = (__bf16*)carve((size_t)BH * DH * T * 2); // v^T [BH,DH,T]
  __bf16* yb  = (__bf16*)carve((size_t)MROWS * C  * 2);  // attn output bf16

  // 1) conversions
  {
    const int n = MROWS * C;
    cvt_f32_bf16_kernel<<<(n + 255) / 256, 256, 0, stream>>>(x, xb, n);
  }
  transpose_f32_bf16_kernel<<<dim3(N3 / 16, C / 16), 256, 0, stream>>>(Wa, WaT, C, N3);
  transpose_f32_bf16_kernel<<<dim3(C / 16,  C / 16), 256, 0, stream>>>(Wp, WpT, C, C);

  // 2) QKV GEMM with head-layout scatter epilogue
  gemm_bf16_wmma_kernel<<<dim3(N3 / 128, MROWS / 128), 256, 0, stream>>>(
      xb, WaT, nullptr, qhB, khB, vtB, MROWS, N3, C, /*mode=*/1);

  // 3) flash attention (1 wave per 16-row q tile)
  flash_attn_wmma_kernel<<<BH * (T / 16), 32, 0, stream>>>(qhB, khB, vtB, yb);

  // 4) output projection GEMM (fp32 out)
  gemm_bf16_wmma_kernel<<<dim3(C / 128, MROWS / 128), 256, 0, stream>>>(
      yb, WpT, out, nullptr, nullptr, nullptr, MROWS, C, C, /*mode=*/0);
}
